// TransformerEncoderLayer_47330539602299
// MI455X (gfx1250) — matrix operand order repaired and checked by reading, not compiled, not run on hardware
//
#include <hip/hip_runtime.h>

#define DEV __device__ __forceinline__

typedef __bf16 bf16_t;
typedef __bf16 v16bf __attribute__((ext_vector_type(16)));
typedef float  v8f   __attribute__((ext_vector_type(8)));
typedef int    v4i   __attribute__((ext_vector_type(4)));

static constexpr int Bb = 8, Ss = 1024, Dd = 768, Hh = 12, Ff = 3072, DHh = 64;
static constexpr int MROWS = Bb * Ss;            // 8192
static constexpr float C_QUAD = 5.0f;
static constexpr float LN_EPS = 1e-5f;

struct __align__(16) bf8 { bf16_t v[8]; };

#define AS1 __attribute__((address_space(1)))
#define AS3 __attribute__((address_space(3)))

#if __has_builtin(__builtin_amdgcn_global_load_async_to_lds_b128) && \
    __has_builtin(__builtin_amdgcn_s_wait_asynccnt)
#define USE_ASYNC_LDS 1
#else
#define USE_ASYNC_LDS 0
#endif

#if USE_ASYNC_LDS
DEV void async_copy16(const bf16_t* g, bf16_t* l) {
    __builtin_amdgcn_global_load_async_to_lds_b128(
        (AS1 v4i*)g, (AS3 v4i*)l, 0, 0);
}
DEV void wait_async0() { __builtin_amdgcn_s_wait_asynccnt(0); }
#endif

DEV bf16_t to_bf(float f) {
    unsigned u = __builtin_bit_cast(unsigned, f);
    u += 0x7FFFu + ((u >> 16) & 1u);             // round-to-nearest-even
    unsigned short s = (unsigned short)(u >> 16);
    return __builtin_bit_cast(bf16_t, s);
}
DEV float bf_to_f(bf16_t b) {
    unsigned short s = __builtin_bit_cast(unsigned short, b);
    unsigned u = ((unsigned)s) << 16;
    return __builtin_bit_cast(float, u);
}

DEV float gelu_f(float x) {
    const float k0 = 0.7978845608028654f;        // sqrt(2/pi)
    float t = tanhf(k0 * (x + 0.044715f * x * x * x));
    return 0.5f * x * (1.0f + t);
}

// ---- WMMA fragment loads from LDS, matching the CDNA5 ISA layouts ----
// A (16x32 bf16, row-major tile, leading dim ld):
//   lane<16  : row=lane,    holds K = 0..7  and 16..23
//   lane>=16 : row=lane-16, holds K = 8..15 and 24..31
DEV v16bf load_a_frag(const bf16_t* p0, int ld) {
    int lane = threadIdx.x & 31;
    const bf16_t* p = p0 + (lane & 15) * ld + ((lane >> 4) << 3);
    v16bf a;
#pragma unroll
    for (int i = 0; i < 8; ++i) a[i] = p[i];
#pragma unroll
    for (int i = 0; i < 8; ++i) a[8 + i] = p[16 + i];
    return a;
}
// B (32x16 bf16) staged transposed in LDS as BsT[n][k]:
//   lane<16  : col=lane,    K = 0..7 and 16..23 ; lane>=16 : col=lane-16, K = 8..15 and 24..31
DEV v16bf load_b_frag(const bf16_t* p0, int ld) {
    int lane = threadIdx.x & 31;
    const bf16_t* p = p0 + (lane & 15) * ld + ((lane >> 4) << 3);
    v16bf b;
#pragma unroll
    for (int i = 0; i < 8; ++i) b[i] = p[i];
#pragma unroll
    for (int i = 0; i < 8; ++i) b[8 + i] = p[16 + i];
    return b;
}

DEV v8f wmma_bf16(v16bf a, v16bf b, v8f c) {
    return __builtin_amdgcn_wmma_f32_16x16x32_bf16(
        false, a, false, b, (short)0, c, false, false);
}

// ---------------- fp32 [K,N] -> bf16 transposed [N,K] ----------------
__global__ void cvt_transpose_kernel(const float* __restrict__ in,
                                     bf16_t* __restrict__ out, int K, int N) {
    int idx = blockIdx.x * blockDim.x + threadIdx.x;
    if (idx < K * N) {
        int k = idx / N, n = idx % N;
        out[(size_t)n * K + k] = to_bf(in[idx]);
    }
}

// ---------------- LayerNorm (row of D=768) -> bf16 ----------------
__global__ void layernorm_bf16_kernel(const float* __restrict__ x,
                                      const float* __restrict__ g,
                                      const float* __restrict__ bch,
                                      bf16_t* __restrict__ out) {
    __shared__ float red[256];
    const int row = blockIdx.x;
    const float* xr = x + (size_t)row * Dd;
    float s = 0.f;
    for (int i = threadIdx.x; i < Dd; i += 256) s += xr[i];
    red[threadIdx.x] = s; __syncthreads();
    for (int off = 128; off > 0; off >>= 1) {
        if (threadIdx.x < off) red[threadIdx.x] += red[threadIdx.x + off];
        __syncthreads();
    }
    const float mu = red[0] / (float)Dd;
    __syncthreads();
    float v = 0.f;
    for (int i = threadIdx.x; i < Dd; i += 256) { float d = xr[i] - mu; v += d * d; }
    red[threadIdx.x] = v; __syncthreads();
    for (int off = 128; off > 0; off >>= 1) {
        if (threadIdx.x < off) red[threadIdx.x] += red[threadIdx.x + off];
        __syncthreads();
    }
    const float rstd = rsqrtf(red[0] / (float)Dd + LN_EPS);
    bf16_t* orow = out + (size_t)row * Dd;
    for (int i = threadIdx.x; i < Dd; i += 256)
        orow[i] = to_bf((xr[i] - mu) * rstd * g[i] + bch[i]);
}

// ---------------- WMMA GEMM: C = act(A@Bt^T + bias) [+ resid] ----------------
// A : [M,K] bf16 row-major.  Bt: [N,K] bf16 row-major (pre-transposed weights).
// 128 threads / 4 waves; block tile 128x128, BK=32; wave tile 64x64 (4x4 frags).
// Double-buffered LDS; async global->LDS DMA when available (ASYNCcnt).
template <bool GELU_ACT, bool RESID, bool OUT_BF16>
__global__ __launch_bounds__(128)
void wmma_gemm_kernel(const bf16_t* __restrict__ A,
                      const bf16_t* __restrict__ Bt,
                      const float* __restrict__ bias,
                      const float* __restrict__ resid,
                      float* __restrict__ outF,
                      bf16_t* __restrict__ outB,
                      int M, int N, int K) {
    __shared__ bf16_t As[2][128 * 32];    // [m][k]
    __shared__ bf16_t BsT[2][128 * 32];   // [n][k]
    const int bm = blockIdx.y * 128, bn = blockIdx.x * 128;
    const int tid = threadIdx.x;
    const int wave = tid >> 5, lane = tid & 31;
    const int wm = (wave >> 1) * 64, wn = (wave & 1) * 64;

    const v8f vzero = {0.f, 0.f, 0.f, 0.f, 0.f, 0.f, 0.f, 0.f};
    v8f acc[4][4];
#pragma unroll
    for (int mi = 0; mi < 4; ++mi)
#pragma unroll
        for (int ni = 0; ni < 4; ++ni) acc[mi][ni] = vzero;

    // stage one 128x32 tile of A and of Bt into LDS buffer `buf`
    auto stage = [&](int buf, int k0) {
#pragma unroll
        for (int i = 0; i < 4; ++i) {
            int v = tid + i * 128;             // 0..511 (segments of 8 bf16)
            int row = v >> 2, c8 = (v & 3) << 3;
            const bf16_t* ga = A  + (size_t)(bm + row) * K + k0 + c8;
            const bf16_t* gb = Bt + (size_t)(bn + row) * K + k0 + c8;
            bf16_t* la = &As[buf][row * 32 + c8];
            bf16_t* lb = &BsT[buf][row * 32 + c8];
#if USE_ASYNC_LDS
            async_copy16(ga, la);
            async_copy16(gb, lb);
#else
            *(bf8*)la = *(const bf8*)ga;
            *(bf8*)lb = *(const bf8*)gb;
#endif
        }
    };
    auto compute = [&](int buf) {
        v16bf af[4], bfr[4];
#pragma unroll
        for (int mi = 0; mi < 4; ++mi)
            af[mi] = load_a_frag(&As[buf][(wm + mi * 16) * 32], 32);
#pragma unroll
        for (int ni = 0; ni < 4; ++ni)
            bfr[ni] = load_b_frag(&BsT[buf][(wn + ni * 16) * 32], 32);
#pragma unroll
        for (int mi = 0; mi < 4; ++mi)
#pragma unroll
            for (int ni = 0; ni < 4; ++ni)
                acc[mi][ni] = wmma_bf16(af[mi], bfr[ni], acc[mi][ni]);
    };

    const int nk = K >> 5;
#if USE_ASYNC_LDS
    stage(0, 0);                               // DMA tile 0
    for (int i = 0; i < nk; ++i) {
        const int cur = i & 1;
        wait_async0();                         // tile i resident in LDS
        __syncthreads();                       // all waves' DMA visible
        if (i + 1 < nk) stage(cur ^ 1, (i + 1) << 5);   // overlap next DMA
        compute(cur);
    }
#else
    for (int i = 0; i < nk; ++i) {
        stage(0, i << 5);
        __syncthreads();
        compute(0);
        __syncthreads();
    }
#endif

    // Epilogue. C/D layout: vgpr r -> row r + 8*(lane>=16), col = lane%16.
    const int colL = lane & 15, hi = lane >> 4;
#pragma unroll
    for (int mi = 0; mi < 4; ++mi) {
#pragma unroll
        for (int ni = 0; ni < 4; ++ni) {
            const int gn = bn + wn + ni * 16 + colL;
            const float bv = bias[gn];
#pragma unroll
            for (int r = 0; r < 8; ++r) {
                const int gm = bm + wm + mi * 16 + hi * 8 + r;
                float val = acc[mi][ni][r] + bv;
                if (GELU_ACT) val = gelu_f(val);
                if (RESID) val += resid[(size_t)gm * N + gn];
                if (OUT_BF16) outB[(size_t)gm * N + gn] = to_bf(val);
                else          outF[(size_t)gm * N + gn] = val;
            }
        }
    }
}

// ---------------- Flash-style quad2 attention ----------------
// grid = (S/64, B*H), 128 threads. p = (s/8 + 5)^2 ; ctx = sum p*v / sum p
__global__ __launch_bounds__(128)
void attn_quad2_kernel(const bf16_t* __restrict__ qkv, bf16_t* __restrict__ ctx) {
    __shared__ bf16_t Qs[64 * 64];    // [q][dh]
    __shared__ bf16_t Ks[64 * 64];    // [key][dh]  (== B^T for q@k^T)
    __shared__ bf16_t VsT[64 * 64];   // [dh][key]  (== B^T for P@V)
    __shared__ bf16_t Ps[64 * 64];    // [q][key]
    __shared__ float denom[64];

    const int bh = blockIdx.y;
    const int b = bh / Hh, h = bh % Hh;
    const int q0 = blockIdx.x * 64;
    const int tid = threadIdx.x, wave = tid >> 5, lane = tid & 31;
    const int wm = (wave >> 1) * 32, wn = (wave & 1) * 32;
    const size_t rs = 3 * Dd;
    const size_t base = (size_t)b * Ss * rs;
    const int qoff = h * DHh, koff = Dd + h * DHh, voff = 2 * Dd + h * DHh;

#pragma unroll
    for (int i = 0; i < 4; ++i) {                          // Q tile (once)
        int v = tid + i * 128;
        int row = v >> 3, c8 = (v & 7) << 3;
        const bf16_t* g = qkv + base + (size_t)(q0 + row) * rs + qoff + c8;
        bf16_t* l = Qs + row * 64 + c8;
#if USE_ASYNC_LDS
        async_copy16(g, l);
#else
        *(bf8*)l = *(const bf8*)g;
#endif
    }
    if (tid < 64) denom[tid] = 0.f;

    const v8f vzero = {0.f, 0.f, 0.f, 0.f, 0.f, 0.f, 0.f, 0.f};
    v8f accc[2][2] = {{vzero, vzero}, {vzero, vzero}};
    __syncthreads();

    const float scale = 0.125f;                            // 1/sqrt(64)
    const int colL = lane & 15, hi = lane >> 4;

    for (int kt = 0; kt < Ss; kt += 64) {
#pragma unroll
        for (int i = 0; i < 4; ++i) {                      // K (async) + V (transpose)
            int v = tid + i * 128;
            int row = v >> 3, c8 = (v & 7) << 3;
            const bf16_t* gk = qkv + base + (size_t)(kt + row) * rs + koff + c8;
#if USE_ASYNC_LDS
            async_copy16(gk, Ks + row * 64 + c8);
#else
            *(bf8*)(Ks + row * 64 + c8) = *(const bf8*)gk;
#endif
            bf8 t = *(const bf8*)(qkv + base + (size_t)(kt + row) * rs + voff + c8);
#pragma unroll
            for (int j = 0; j < 8; ++j) VsT[(c8 + j) * 64 + row] = t.v[j];
        }
#if USE_ASYNC_LDS
        wait_async0();                                     // Q (iter 0) + K ready
#endif
        __syncthreads();

        // scores subtile: q rows [wm,wm+32), keys [wn,wn+32), K-dim = DH = 64
        v8f sacc[2][2] = {{vzero, vzero}, {vzero, vzero}};
#pragma unroll
        for (int ks = 0; ks < 64; ks += 32) {
            v16bf a0 = load_a_frag(Qs + (wm + 0) * 64 + ks, 64);
            v16bf a1 = load_a_frag(Qs + (wm + 16) * 64 + ks, 64);
            v16bf b0 = load_b_frag(Ks + (wn + 0) * 64 + ks, 64);
            v16bf b1 = load_b_frag(Ks + (wn + 16) * 64 + ks, 64);
            sacc[0][0] = wmma_bf16(a0, b0, sacc[0][0]);
            sacc[0][1] = wmma_bf16(a0, b1, sacc[0][1]);
            sacc[1][0] = wmma_bf16(a1, b0, sacc[1][0]);
            sacc[1][1] = wmma_bf16(a1, b1, sacc[1][1]);
        }
#pragma unroll
        for (int mi = 0; mi < 2; ++mi)                     // quad2 numerator -> Ps
#pragma unroll
            for (int ni = 0; ni < 2; ++ni)
#pragma unroll
                for (int r = 0; r < 8; ++r) {
                    float sc = sacc[mi][ni][r] * scale + C_QUAD;
                    Ps[(wm + mi * 16 + hi * 8 + r) * 64 + wn + ni * 16 + colL] =
                        to_bf(sc * sc);
                }
        __syncthreads();

        if (tid < 64) {                                    // denominator rows
            float sum = 0.f;
            for (int j = 0; j < 64; ++j) sum += bf_to_f(Ps[tid * 64 + j]);
            denom[tid] += sum;
        }
        // ctx += P @ V   (K-dim = 64 keys)
#pragma unroll
        for (int ks = 0; ks < 64; ks += 32) {
            v16bf a0 = load_a_frag(Ps + (wm + 0) * 64 + ks, 64);
            v16bf a1 = load_a_frag(Ps + (wm + 16) * 64 + ks, 64);
            v16bf b0 = load_b_frag(VsT + (wn + 0) * 64 + ks, 64);
            v16bf b1 = load_b_frag(VsT + (wn + 16) * 64 + ks, 64);
            accc[0][0] = wmma_bf16(a0, b0, accc[0][0]);
            accc[0][1] = wmma_bf16(a0, b1, accc[0][1]);
            accc[1][0] = wmma_bf16(a1, b0, accc[1][0]);
            accc[1][1] = wmma_bf16(a1, b1, accc[1][1]);
        }
        __syncthreads();
    }

#pragma unroll
    for (int mi = 0; mi < 2; ++mi)
#pragma unroll
        for (int ni = 0; ni < 2; ++ni)
#pragma unroll
            for (int r = 0; r < 8; ++r) {
                int qrow = wm + mi * 16 + hi * 8 + r;
                int dcol = wn + ni * 16 + colL;
                float val = accc[mi][ni][r] / denom[qrow];
                ctx[(size_t)(b * Ss + q0 + qrow) * Dd + h * DHh + dcol] = to_bf(val);
            }
}

// ---------------- host ----------------
extern "C" void kernel_launch(void* const* d_in, const int* in_sizes, int n_in,
                              void* d_out, int out_size, void* d_ws, size_t ws_size,
                              hipStream_t stream) {
    (void)in_sizes; (void)n_in; (void)out_size; (void)ws_size;
    const float* x    = (const float*)d_in[0];
    const float* wqkv = (const float*)d_in[1];
    const float* bqkv = (const float*)d_in[2];
    const float* wo   = (const float*)d_in[3];
    const float* bo   = (const float*)d_in[4];
    const float* ln1g = (const float*)d_in[5];
    const float* ln1b = (const float*)d_in[6];
    const float* w1   = (const float*)d_in[7];
    const float* b1   = (const float*)d_in[8];
    const float* w2   = (const float*)d_in[9];
    const float* b2   = (const float*)d_in[10];
    const float* ln2g = (const float*)d_in[11];
    const float* ln2b = (const float*)d_in[12];
    float* out = (float*)d_out;

    char* ws = (char*)d_ws;
    size_t off = 0;
    auto carve = [&](size_t bytes) -> char* {
        char* p = ws + off;
        off += (bytes + 255) & ~(size_t)255;
        return p;
    };
    bf16_t* wqkvT = (bf16_t*)carve((size_t)Dd * 3 * Dd * 2);   // [3D][D]
    bf16_t* woT   = (bf16_t*)carve((size_t)Dd * Dd * 2);       // [D][D]
    bf16_t* w1T   = (bf16_t*)carve((size_t)Dd * Ff * 2);       // [F][D]
    bf16_t* w2T   = (bf16_t*)carve((size_t)Ff * Dd * 2);       // [D][F]
    bf16_t* hb    = (bf16_t*)carve((size_t)MROWS * Dd * 2);    // LN out (reused)
    bf16_t* qkv_b = (bf16_t*)carve((size_t)MROWS * 3 * Dd * 2);
    bf16_t* ctx_b = (bf16_t*)carve((size_t)MROWS * Dd * 2);
    float*  x1    = (float*) carve((size_t)MROWS * Dd * 4);
    bf16_t* ff1   = (bf16_t*)carve((size_t)MROWS * Ff * 2);

    auto cvtT = [&](const float* src, bf16_t* dst, int K, int N) {
        cvt_transpose_kernel<<<(K * N + 255) / 256, 256, 0, stream>>>(src, dst, K, N);
    };
    cvtT(wqkv, wqkvT, Dd, 3 * Dd);
    cvtT(wo,   woT,   Dd, Dd);
    cvtT(w1,   w1T,   Dd, Ff);
    cvtT(w2,   w2T,   Ff, Dd);

    // LN1
    layernorm_bf16_kernel<<<MROWS, 256, 0, stream>>>(x, ln1g, ln1b, hb);
    // QKV = hb @ wqkv + bqkv  -> bf16
    {
        dim3 g(3 * Dd / 128, MROWS / 128);
        wmma_gemm_kernel<false, false, true><<<g, 128, 0, stream>>>(
            hb, wqkvT, bqkv, nullptr, nullptr, qkv_b, MROWS, 3 * Dd, Dd);
    }
    // attention (quad2)
    {
        dim3 g(Ss / 64, Bb * Hh);
        attn_quad2_kernel<<<g, 128, 0, stream>>>(qkv_b, ctx_b);
    }
    // x1 = x + ctx @ wo + bo  -> fp32
    {
        dim3 g(Dd / 128, MROWS / 128);
        wmma_gemm_kernel<false, true, false><<<g, 128, 0, stream>>>(
            ctx_b, woT, bo, x, x1, nullptr, MROWS, Dd, Dd);
    }
    // LN2
    layernorm_bf16_kernel<<<MROWS, 256, 0, stream>>>(x1, ln2g, ln2b, hb);
    // ff1 = gelu(hb @ w1 + b1) -> bf16
    {
        dim3 g(Ff / 128, MROWS / 128);
        wmma_gemm_kernel<true, false, true><<<g, 128, 0, stream>>>(
            hb, w1T, b1, nullptr, nullptr, ff1, MROWS, Ff, Dd);
    }
    // out = x1 + ff1 @ w2 + b2 -> fp32
    {
        dim3 g(Dd / 128, MROWS / 128);
        wmma_gemm_kernel<false, true, false><<<g, 128, 0, stream>>>(
            ff1, w2T, b2, x1, out, nullptr, MROWS, Dd, Ff);
    }
}
